// SegmentwiseInterpretableMultiHeadAttention_16054587752654
// MI455X (gfx1250) — matrix operand order repaired and checked by reading, not compiled, not run on hardware
//
#include <hip/hip_runtime.h>

// ---------------- shapes (fixed by the reference) ----------------
#define BB    2
#define LQ    1024
#define LKV   2048
#define DD    1024
#define HH    16
#define DK    64
#define DV    64
#define SEG   256
#define NSEG  8          // LKV / SEG
#define LL    8192       // NSEG * LQ

typedef _Float16 v8h  __attribute__((ext_vector_type(8)));
typedef _Float16 v16h __attribute__((ext_vector_type(16)));
typedef float    v8f  __attribute__((ext_vector_type(8)));
typedef int      v4i  __attribute__((ext_vector_type(4)));

#if defined(__has_builtin)
#if __has_builtin(__builtin_amdgcn_global_load_async_to_lds_b128) && \
    __has_builtin(__builtin_amdgcn_s_wait_asynccnt)
#define HAVE_ASYNC_COPY 1
#endif
#endif

static __device__ __forceinline__ v8f zero_v8f() {
  v8f z;
#pragma unroll
  for (int i = 0; i < 8; ++i) z[i] = 0.0f;
  return z;
}

// A-fragment (16x32 f16): lane L holds row L%16.
// slots 0..7  -> K = k0 + 8*(L/16) + i ; slots 8..15 -> +16
static __device__ __forceinline__ v16h load_a_frag(const _Float16* rowPtr, int k0, int h16) {
  v8h lo = *(const v8h*)(rowPtr + k0 + 8 * h16);
  v8h hi = *(const v8h*)(rowPtr + k0 + 16 + 8 * h16);
  v16h a;
#pragma unroll
  for (int i = 0; i < 8; ++i) { a[i] = lo[i]; a[i + 8] = hi[i]; }
  return a;
}

// B-fragment (32x16 f16): lane L holds column L%16; slots 0..7 -> K = k0 + 8*(L/16) + i ; slots 8..15 -> +16
static __device__ __forceinline__ v16h load_b_frag(const _Float16* colPtr, int k0, int h16) {
  v8h lo = *(const v8h*)(colPtr + k0 + 8 * h16);
  v8h hi = *(const v8h*)(colPtr + k0 + 16 + 8 * h16);
  v16h b;
#pragma unroll
  for (int i = 0; i < 8; ++i) { b[i] = lo[i]; b[i + 8] = hi[i]; }
  return b;
}

#define WMMA(a, b, c) __builtin_amdgcn_wmma_f32_16x16x32_f16(false, (a), false, (b), (short)0, (c), false, false)

// ---------------- prep kernels ----------------
__global__ void cvt_f16_kernel(const float* __restrict__ src, _Float16* __restrict__ dst, int n) {
  int i = blockIdx.x * 256 + threadIdx.x;
  if (i < n) dst[i] = (_Float16)src[i];
}

// W[h][r][c] (f32) -> WT[h][c][r] (f16)
__global__ void transpose_w_kernel(const float* __restrict__ W, _Float16* __restrict__ WT,
                                   int Hn, int R, int C) {
  int i = blockIdx.x * 256 + threadIdx.x;
  int total = Hn * R * C;
  if (i >= total) return;
  int h = i / (R * C);
  int rem = i % (R * C);
  int c = rem / R;
  int r = rem % R;
  WT[i] = (_Float16)W[((size_t)h * R + r) * C + c];
}

// ---------------- projection GEMM: 64x64 register block per wave ----------------
// BT layout: [N][K] (each B column contiguous along K).
// mode 0: out[((b*HH+h)*Lrow + rr)*DK + kk]  (per-head layout, n = h*64+kk)
// mode 1: out[(b*DV + n)*Lrow + s]           (vpT layout)
__global__ __launch_bounds__(256) void proj_gemm(const _Float16* __restrict__ A,
                                                 const _Float16* __restrict__ BT,
                                                 const float* __restrict__ bias,
                                                 _Float16* __restrict__ out,
                                                 int M, int N, int K, int mode, int Lrow) {
  const int wave = threadIdx.x >> 5, lane = threadIdx.x & 31;
  const int h16 = lane >> 4, l16 = lane & 15;
  const int bt_n = N >> 6;
  const int blk = blockIdx.x * 8 + wave;
  if (blk >= (M >> 6) * bt_n) return;
  const int m0 = (blk / bt_n) << 6;
  const int n0 = (blk % bt_n) << 6;

  v8f acc[16];
#pragma unroll
  for (int t = 0; t < 16; ++t) acc[t] = zero_v8f();

#pragma unroll 2
  for (int k0 = 0; k0 < K; k0 += 32) {
    v16h af[4], bf[4];
#pragma unroll
    for (int i = 0; i < 4; ++i)
      af[i] = load_a_frag(A + (size_t)(m0 + 16 * i + l16) * K, k0, h16);
#pragma unroll
    for (int j = 0; j < 4; ++j)
      bf[j] = load_b_frag(BT + (size_t)(n0 + 16 * j + l16) * K, k0, h16);
#pragma unroll
    for (int i = 0; i < 4; ++i)
#pragma unroll
      for (int j = 0; j < 4; ++j)
        acc[i * 4 + j] = WMMA(af[i], bf[j], acc[i * 4 + j]);
  }

#pragma unroll
  for (int j = 0; j < 4; ++j) {
    const int n = n0 + 16 * j + l16;
    const float bb = bias ? bias[n] : 0.0f;
#pragma unroll
    for (int i = 0; i < 4; ++i)
#pragma unroll
      for (int r = 0; r < 8; ++r) {
        int m = m0 + 16 * i + 8 * h16 + r;
        float val = acc[i * 4 + j][r] + bb;
        size_t idx;
        if (mode == 0) {
          int bi = m / Lrow, rr = m % Lrow;
          int hh = n >> 6, kk = n & 63;
          idx = (((size_t)(bi * HH + hh)) * Lrow + rr) * DK + kk;
        } else {
          int bi = m / Lrow, s = m % Lrow;
          idx = ((size_t)bi * DV + n) * Lrow + s;
        }
        out[idx] = (_Float16)val;
      }
  }
}

// ---------------- attention: per (b,h,seg), wave = 16 q-rows x full S ----------------
#define AT_WAVES  4
#define KPITCH    72    // 64 + 8 halves pad: 36*s mod 64 -> 16 distinct banks
#define LPITCH    264   // 256 + 8 halves pad for probs staging

__global__ __launch_bounds__(128) void attn_kernel(const _Float16* __restrict__ qs,
                                                   const _Float16* __restrict__ kp,
                                                   const _Float16* __restrict__ vpT,
                                                   float* __restrict__ attn_out,
                                                   float* __restrict__ head) {
  // 36 KB, reused: phase 1 = K-segment tile [256][72]; phase 2 = probs [4][16][264]
  __shared__ __align__(16) _Float16 smem[SEG * KPITCH];
  const int wave = threadIdx.x >> 5, lane = threadIdx.x & 31;
  const int h16 = lane >> 4, l16 = lane & 15;
  const int b = blockIdx.z >> 4, h = blockIdx.z & 15;
  const int seg = blockIdx.y;
  const int q0 = blockIdx.x * (AT_WAVES * 16) + wave * 16;

  const _Float16* qrow  = qs + (((size_t)(b * HH + h)) * LQ + (q0 + l16)) * DK;
  const _Float16* kbase = kp + (((size_t)(b * HH + h)) * LKV + seg * SEG) * DK;

  // ---- stage K-segment tile (256 x 64 f16) into LDS, once per block ----
  // 16-byte chunks: global row = 8 chunks, LDS row = 9 chunks (KPITCH=72 halves)
#ifdef HAVE_ASYNC_COPY
  {
    __attribute__((address_space(1))) v4i* gsrc =
        (__attribute__((address_space(1))) v4i*)kbase;
    __attribute__((address_space(3))) v4i* ldst =
        (__attribute__((address_space(3))) v4i*)smem;
    for (int c = threadIdx.x; c < SEG * 8; c += 128) {
      int s = c >> 3, cc = c & 7;
      __builtin_amdgcn_global_load_async_to_lds_b128(gsrc + (s * 8 + cc),
                                                     ldst + (s * 9 + cc), 0, 0);
    }
    __builtin_amdgcn_s_wait_asynccnt(0);
  }
#else
  for (int c = threadIdx.x; c < SEG * 8; c += 128) {
    int s = c >> 3, cc = c & 7;
    *(v8h*)&smem[s * KPITCH + cc * 8] = *(const v8h*)(kbase + (size_t)s * DK + cc * 8);
  }
#endif
  __syncthreads();

  // warm V columns we will need after softmax
#pragma unroll
  for (int nt = 0; nt < 4; ++nt)
    __builtin_prefetch(vpT + ((size_t)b * DV + nt * 16 + l16) * LKV + seg * SEG, 0, 1);

  // ---- scores = qs * kp^T : 16 tiles of 16x16, K = dk = 64 (B-frags from LDS) ----
  v8f acc[16];
#pragma unroll
  for (int t = 0; t < 16; ++t) acc[t] = zero_v8f();

  v16h a0 = load_a_frag(qrow, 0, h16);
  v16h a1 = load_a_frag(qrow, 32, h16);
#pragma unroll
  for (int t = 0; t < 16; ++t) {
    const _Float16* kcol = smem + (t * 16 + l16) * KPITCH;
    acc[t] = WMMA(a0, load_b_frag(kcol, 0, h16), acc[t]);
    acc[t] = WMMA(a1, load_b_frag(kcol, 32, h16), acc[t]);
  }

  // ---- scale, causal mask (s > q), softmax over s in [0,256) ----
  float inv[8];
#pragma unroll
  for (int r = 0; r < 8; ++r) {
    int q_row = q0 + 8 * h16 + r;
    float mx = -3.0e38f;
#pragma unroll
    for (int t = 0; t < 16; ++t) {
      float vsc = acc[t][r] * 0.125f;           // 1/sqrt(dk)
      if (t * 16 + l16 > q_row) vsc = -1.0e9f;  // same mask for every segment
      acc[t][r] = vsc;
      mx = fmaxf(mx, vsc);
    }
#pragma unroll
    for (int off = 1; off < 16; off <<= 1) mx = fmaxf(mx, __shfl_xor(mx, off, 32));
    float sum = 0.0f;
#pragma unroll
    for (int t = 0; t < 16; ++t) {
      float e = __expf(acc[t][r] - mx);
      acc[t][r] = e;
      sum += e;
    }
#pragma unroll
    for (int off = 1; off < 16; off <<= 1) sum += __shfl_xor(sum, off, 32);
    inv[r] = 1.0f / sum;
  }

  __syncthreads();  // everyone done reading the K tile; LDS is re-purposed for probs

  // ---- write attn probs (fp32) + stage f16 probs to LDS for PV A-fragments ----
  float* aout = attn_out + ((((size_t)b * LL) + (size_t)seg * LQ) * HH + h) * SEG;
  _Float16* lw = smem + wave * 16 * LPITCH;
#pragma unroll
  for (int r = 0; r < 8; ++r) {
    int mrow = 8 * h16 + r;
    int q_row = q0 + mrow;
    float iv = inv[r];
    size_t rowoff = (size_t)q_row * (HH * SEG);
#pragma unroll
    for (int t = 0; t < 16; ++t) {
      int s = t * 16 + l16;
      float p = acc[t][r] * iv;
      aout[rowoff + s] = p;
      lw[mrow * LPITCH + s] = (_Float16)p;
    }
  }
  __syncthreads();

  // ---- out = P[16x256] * vp[256x64] : 4 n-tiles, 8 K-steps ----
  v8f o[4];
#pragma unroll
  for (int nt = 0; nt < 4; ++nt) o[nt] = zero_v8f();
#pragma unroll
  for (int ks = 0; ks < 8; ++ks) {
    v16h ap = load_a_frag(lw + l16 * LPITCH, ks * 32, h16);
#pragma unroll
    for (int nt = 0; nt < 4; ++nt) {
      const _Float16* vcol = vpT + ((size_t)b * DV + nt * 16 + l16) * LKV + seg * SEG;
      o[nt] = WMMA(ap, load_b_frag(vcol, ks * 32, h16), o[nt]);
    }
  }
#pragma unroll
  for (int nt = 0; nt < 4; ++nt)
#pragma unroll
    for (int r = 0; r < 8; ++r) {
      int q_row = q0 + 8 * h16 + r;
      size_t idx = ((((size_t)b * LL) + (size_t)seg * LQ + q_row) * HH + h) * DV + nt * 16 + l16;
      head[idx] = o[nt][r];
    }
}

// ---------------- mean over heads (deterministic, no atomics) ----------------
__global__ void mean_head_kernel(const float* __restrict__ head, _Float16* __restrict__ hm) {
  int i = blockIdx.x * 256 + threadIdx.x;      // B*L*dv = 1048576
  if (i >= BB * LL * DV) return;
  int v = i & 63;
  size_t bl = (size_t)(i >> 6);
  float s = 0.0f;
#pragma unroll
  for (int h = 0; h < HH; ++h) s += head[(bl * HH + h) * DV + v];
  hm[i] = (_Float16)(s * (1.0f / HH));
}

// ---------------- outputs = hm[16384x64] @ Wh[64x1024], 64x64 block/wave ----------------
__global__ __launch_bounds__(256) void final_gemm(const _Float16* __restrict__ A,
                                                  const _Float16* __restrict__ BT,
                                                  float* __restrict__ out) {
  const int wave = threadIdx.x >> 5, lane = threadIdx.x & 31;
  const int h16 = lane >> 4, l16 = lane & 15;
  const int blk = blockIdx.x * 8 + wave;       // bt_n = 1024/64 = 16
  const int m0 = (blk >> 4) << 6;
  const int n0 = (blk & 15) << 6;

  v8f acc[16];
#pragma unroll
  for (int t = 0; t < 16; ++t) acc[t] = zero_v8f();

#pragma unroll
  for (int k0 = 0; k0 < DK; k0 += 32) {
    v16h af[4], bf[4];
#pragma unroll
    for (int i = 0; i < 4; ++i)
      af[i] = load_a_frag(A + (size_t)(m0 + 16 * i + l16) * DK, k0, h16);
#pragma unroll
    for (int j = 0; j < 4; ++j)
      bf[j] = load_b_frag(BT + (size_t)(n0 + 16 * j + l16) * DK, k0, h16);
#pragma unroll
    for (int i = 0; i < 4; ++i)
#pragma unroll
      for (int j = 0; j < 4; ++j)
        acc[i * 4 + j] = WMMA(af[i], bf[j], acc[i * 4 + j]);
  }

#pragma unroll
  for (int i = 0; i < 4; ++i)
#pragma unroll
    for (int j = 0; j < 4; ++j)
#pragma unroll
      for (int r = 0; r < 8; ++r)
        out[(size_t)(m0 + 16 * i + 8 * h16 + r) * DD + n0 + 16 * j + l16] = acc[i * 4 + j][r];
}

// ---------------- host orchestration ----------------
extern "C" void kernel_launch(void* const* d_in, const int* in_sizes, int n_in,
                              void* d_out, int out_size, void* d_ws, size_t ws_size,
                              hipStream_t stream) {
  (void)in_sizes; (void)n_in; (void)out_size; (void)ws_size;

  const float* q  = (const float*)d_in[0];
  const float* k  = (const float*)d_in[1];
  const float* v  = (const float*)d_in[2];
  const float* Wq = (const float*)d_in[3];
  const float* bq = (const float*)d_in[4];
  const float* Wk = (const float*)d_in[5];
  const float* bk = (const float*)d_in[6];
  const float* Wv = (const float*)d_in[7];
  const float* bv = (const float*)d_in[8];
  const float* Wh = (const float*)d_in[9];

  char* ws = (char*)d_ws;
  size_t off = 0;
  auto carve = [&](size_t bytes) -> char* {
    char* p = ws + off;
    off += (bytes + 255) & ~(size_t)255;
    return p;
  };

  _Float16* q16  = (_Float16*)carve((size_t)BB * LQ * DD * 2);         // 4 MB
  _Float16* k16  = (_Float16*)carve((size_t)BB * LKV * DD * 2);        // 8 MB
  _Float16* v16  = (_Float16*)carve((size_t)BB * LKV * DD * 2);        // 8 MB
  _Float16* WqT  = (_Float16*)carve((size_t)HH * DK * DD * 2);         // 2 MB
  _Float16* WkT  = (_Float16*)carve((size_t)HH * DK * DD * 2);         // 2 MB
  _Float16* WvT  = (_Float16*)carve((size_t)DK * DD * 2);              // 128 KB
  _Float16* WhT  = (_Float16*)carve((size_t)DD * DK * 2);              // 128 KB
  _Float16* qs16 = (_Float16*)carve((size_t)BB * HH * LQ * DK * 2);    // 4 MB
  _Float16* kp16 = (_Float16*)carve((size_t)BB * HH * LKV * DK * 2);   // 8 MB
  _Float16* vpT  = (_Float16*)carve((size_t)BB * DV * LKV * 2);        // 512 KB
  float*    head = (float*)   carve((size_t)BB * LL * HH * DV * 4);    // 64 MB
  _Float16* hm16 = (_Float16*)carve((size_t)BB * LL * DV * 2);         // 2 MB

  // 1) fp32 -> fp16 activations
  cvt_f16_kernel<<<(BB * LQ * DD + 255) / 256, 256, 0, stream>>>(q, q16, BB * LQ * DD);
  cvt_f16_kernel<<<(BB * LKV * DD + 255) / 256, 256, 0, stream>>>(k, k16, BB * LKV * DD);
  cvt_f16_kernel<<<(BB * LKV * DD + 255) / 256, 256, 0, stream>>>(v, v16, BB * LKV * DD);

  // 2) weight transpose+convert (B columns contiguous along K)
  transpose_w_kernel<<<(HH * DD * DK + 255) / 256, 256, 0, stream>>>(Wq, WqT, HH, DD, DK);
  transpose_w_kernel<<<(HH * DD * DK + 255) / 256, 256, 0, stream>>>(Wk, WkT, HH, DD, DK);
  transpose_w_kernel<<<(DD * DK + 255) / 256, 256, 0, stream>>>(Wv, WvT, 1, DD, DK);
  transpose_w_kernel<<<(DK * DD + 255) / 256, 256, 0, stream>>>(Wh, WhT, 1, DK, DD);

  // 3) projections (WMMA GEMMs, 64x64 block per wave)
  proj_gemm<<<(BB * LQ / 64) * (HH * DK / 64) / 8, 256, 0, stream>>>(
      q16, WqT, bq, qs16, BB * LQ, HH * DK, DD, 0, LQ);
  proj_gemm<<<(BB * LKV / 64) * (HH * DK / 64) / 8, 256, 0, stream>>>(
      k16, WkT, bk, kp16, BB * LKV, HH * DK, DD, 0, LKV);
  proj_gemm<<<(BB * LKV / 64) * (DV / 64) / 8, 256, 0, stream>>>(
      v16, WvT, bv, vpT, BB * LKV, DV, DD, 1, LKV);

  // 4) attention: QK^T -> mask -> softmax -> attn_out + P*V -> head
  float* outp = (float*)d_out;                       // outputs [B,L,D]
  float* attn = outp + (size_t)BB * LL * DD;         // attn_out [B,L,H,S]
  attn_kernel<<<dim3(LQ / (AT_WAVES * 16), NSEG, BB * HH), AT_WAVES * 32, 0, stream>>>(
      qs16, kp16, vpT, attn, head);

  // 5) head mean over H, then final WMMA GEMM into outputs
  mean_head_kernel<<<(BB * LL * DV + 255) / 256, 256, 0, stream>>>(head, hm16);
  final_gemm<<<(BB * LL / 64) * (DD / 64) / 8, 256, 0, stream>>>(hm16, WhT, outp);
}
